// GroupedQueryAttention_27230092657073
// MI455X (gfx1250) — matrix operand order repaired and checked by reading, not compiled, not run on hardware
//
#include <hip/hip_runtime.h>

#if defined(__has_builtin)
#  if __has_builtin(__builtin_amdgcn_tensor_load_to_lds)
#    define HAVE_TDM 1
#  endif
#endif
#if __has_include(<hip/amd_detail/amd_gfx1250_TDM.h>)
#  define TDM_ARGS6 1
#endif

// ---------------- problem constants ----------------
static constexpr int B    = 4;
static constexpr int S    = 2048;
static constexpr int E    = 1024;
static constexpr int H    = 4;      // kv heads
static constexpr int D    = 64;     // head dim
static constexpr int S16  = S / 16; // 128 q-tiles per (b,h)

typedef __bf16 bf16_t;
typedef bf16_t v16bf __attribute__((ext_vector_type(16)));
typedef float  v8f   __attribute__((ext_vector_type(8)));
typedef unsigned int u32x4 __attribute__((ext_vector_type(4)));
typedef int          i32x4 __attribute__((ext_vector_type(4)));
typedef int          i32x8 __attribute__((ext_vector_type(8)));

union FragB16 {
  v16bf v;
  unsigned int d[8];
  unsigned short u[16];
};

// ---------------- small helpers ----------------
static __device__ __forceinline__ unsigned short f2bf(float f) {
  unsigned int u = __float_as_uint(f);
  u = u + 0x7FFFu + ((u >> 16) & 1u);   // round-to-nearest-even
  return (unsigned short)(u >> 16);
}
static __device__ __forceinline__ float bf2f(unsigned short h) {
  return __uint_as_float(((unsigned int)h) << 16);
}
static __device__ __forceinline__ unsigned int pack2(float a, float b) {
  return (unsigned int)f2bf(a) | ((unsigned int)f2bf(b) << 16);
}
static __device__ __forceinline__ v8f zero8() {
  v8f z;
#pragma unroll
  for (int i = 0; i < 8; ++i) z[i] = 0.0f;
  return z;
}
static __device__ __forceinline__ v8f wmma_bf16(const FragB16& a, const FragB16& b, v8f c) {
  return __builtin_amdgcn_wmma_f32_16x16x32_bf16(false, a.v, false, b.v,
                                                 (short)0, c, false, false);
}

// A fragment (16x32 bf16, M x K): lane l<16 row M=l, v0..3 K=hi*16+{0..7}, v4..7 K=hi*16+8+{0..7}
static __device__ __forceinline__ void loadA(FragB16& f, const unsigned short* base,
                                             int ld, int row0, int k0, int lane) {
  int l = lane & 15, hi = lane >> 4;
  const unsigned short* p = base + (size_t)(row0 + l) * ld + k0 + hi * 16;
#pragma unroll
  for (int j = 0; j < 4; ++j) f.d[j] = *(const unsigned int*)(p + 2 * j);
  p += 8;
#pragma unroll
  for (int j = 0; j < 4; ++j) f.d[4 + j] = *(const unsigned int*)(p + 2 * j);
}

// B fragment from pre-swizzled storage: 32 lanes x 8 dwords, lane-major (32B/lane)
static __device__ __forceinline__ void loadBsw(FragB16& f, const unsigned int* frag, int lane) {
  const uint4* p = (const uint4*)frag + (size_t)lane * 2;
  uint4 x = p[0], y = p[1];
  f.d[0] = x.x; f.d[1] = x.y; f.d[2] = x.z; f.d[3] = x.w;
  f.d[4] = y.x; f.d[5] = y.y; f.d[6] = y.z; f.d[7] = y.w;
}

#if HAVE_TDM
// 1-D TDM copy of 8192 bytes global -> LDS (1024 elements of 8B), per CDNA5 D# layout.
static __device__ __forceinline__ void tdm_load_8k(unsigned lds_bytes, const void* gptr) {
  unsigned long long ga = (unsigned long long)gptr;
  u32x4 g0;
  g0[0] = 1u;                                        // count=1 (user descriptor)
  g0[1] = lds_bytes;                                 // lds_addr (bytes)
  g0[2] = (unsigned)(ga & 0xFFFFFFFFu);              // global_addr[31:0]
  g0[3] = (unsigned)((ga >> 32) & 0x01FFFFFFu)       // global_addr[56:32]
        | (2u << 30);                                // type = 2 ("image")
  const unsigned n8 = 8192u / 8u;                    // 1024 x 8B
  i32x8 g1;
  g1[0] = (int)(3u << 16);                           // workgroup_mask=0, data_size=8B
  g1[1] = (int)((n8 & 0xFFFFu) << 16);               // tensor_dim0[15:0]
  g1[2] = (int)((n8 >> 16) | (1u << 16));            // tensor_dim0[31:16] | tensor_dim1[15:0]=1
  g1[3] = (int)(n8 << 16);                           // tensor_dim1[31:16]=0 | tile_dim0=1024
  g1[4] = 1;                                         // tile_dim1=1, tile_dim2=0
  g1[5] = (int)n8;                                   // tensor_dim0_stride[31:0]
  g1[6] = 0;                                         // stride0[47:32], stride1[15:0]
  g1[7] = 0;
  i32x4 z4 = {0, 0, 0, 0};
#if defined(TDM_ARGS6)
  i32x8 z8 = {0, 0, 0, 0, 0, 0, 0, 0};
  __builtin_amdgcn_tensor_load_to_lds(g0, g1, z4, z4, z8, 0);
#else
  __builtin_amdgcn_tensor_load_to_lds(g0, g1, z4, z4, 0);
#endif
}
#endif

// ---------------- prep kernels ----------------
__global__ void k_convert_bf16(const float* __restrict__ src, unsigned short* __restrict__ dst, int n) {
  int i = (blockIdx.x * blockDim.x + threadIdx.x) * 4;
  if (i >= n) return;
  float4 v = *(const float4*)(src + i);
  *(uint2*)(dst + i) = make_uint2(pack2(v.x, v.y), pack2(v.z, v.w));
}

// WqSum swizzled: frag f = h*128 + kt*4 + nt  (kt over E/32=32, nt over D/16=4)
__global__ void k_prep_wq(const float* __restrict__ Wq, unsigned int* __restrict__ sw) {
  int t = blockIdx.x * 256 + threadIdx.x;            // 131072
  int elem = t & 255, f = t >> 8;
  int lane = elem >> 3, j = elem & 7;
  int n0 = lane & 15, khalf = lane >> 4;
  int nt = f & 3, kt = (f >> 2) & 31, h = f >> 7;
  int e = kt * 32 + khalf * 16 + 2 * j;
  int d = nt * 16 + n0;
  float v0 = 0.f, v1 = 0.f;
#pragma unroll
  for (int g = 0; g < 4; ++g) {
    v0 += Wq[((size_t)e       * 16 + h * 4 + g) * 64 + d];
    v1 += Wq[((size_t)(e + 1) * 16 + h * 4 + g) * 64 + d];
  }
  sw[t] = pack2(v0, v1);
}

__global__ void k_prep_wkv(const float* __restrict__ W, unsigned int* __restrict__ sw) {
  int t = blockIdx.x * 256 + threadIdx.x;            // 131072
  int elem = t & 255, f = t >> 8;
  int lane = elem >> 3, j = elem & 7;
  int n0 = lane & 15, khalf = lane >> 4;
  int nt = f & 3, kt = (f >> 2) & 31, h = f >> 7;
  int e = kt * 32 + khalf * 16 + 2 * j;
  int d = nt * 16 + n0;
  float v0 = W[((size_t)e       * 4 + h) * 64 + d];
  float v1 = W[((size_t)(e + 1) * 4 + h) * 64 + d];
  sw[t] = pack2(v0, v1);
}

// Wo (64x1024) swizzled: frag f = kt*64 + nt
__global__ void k_prep_wo(const float* __restrict__ Wo, unsigned int* __restrict__ sw) {
  int t = blockIdx.x * 256 + threadIdx.x;            // 32768
  int elem = t & 255, f = t >> 8;
  int lane = elem >> 3, j = elem & 7;
  int n0 = lane & 15, khalf = lane >> 4;
  int nt = f & 63, kt = f >> 6;
  int d = kt * 32 + khalf * 16 + 2 * j;
  int e = nt * 16 + n0;
  sw[t] = pack2(Wo[(size_t)d * 1024 + e], Wo[(size_t)(d + 1) * 1024 + e]);
}

// ---------------- projection + RoPE ----------------
__global__ void __launch_bounds__(128) k_proj(
    const unsigned short* __restrict__ qbf, const unsigned short* __restrict__ kvbf,
    const unsigned int* __restrict__ wqsw, const unsigned int* __restrict__ wksw,
    const unsigned int* __restrict__ wvsw,
    unsigned short* __restrict__ qrope, unsigned short* __restrict__ krope,
    unsigned short* __restrict__ vbf) {
  int lane = threadIdx.x & 31;
  int W = blockIdx.x * 4 + (threadIdx.x >> 5);
  if (W >= 3 * B * H * S16) return;
  int typ = W / (B * H * S16);
  int rem = W % (B * H * S16);
  int bh = rem / S16, st = rem % S16;
  int b = bh / H;

  const unsigned short* A = (typ == 0 ? qbf : kvbf) + (size_t)b * S * E;
  const unsigned int* Bsw = (typ == 0 ? wqsw : (typ == 1 ? wksw : wvsw)) + (size_t)(bh % H) * 32768;

  v8f acc[4];
#pragma unroll
  for (int nt = 0; nt < 4; ++nt) acc[nt] = zero8();

  for (int kt = 0; kt < 32; ++kt) {
    FragB16 a;
    loadA(a, A, E, st * 16, kt * 32, lane);
#pragma unroll
    for (int nt = 0; nt < 4; ++nt) {
      FragB16 bfr;
      loadBsw(bfr, Bsw + ((size_t)kt * 4 + nt) * 256, lane);
      acc[nt] = wmma_bf16(a, bfr, acc[nt]);
    }
  }

  int n0 = lane & 15, hi = lane >> 4;
  unsigned short* dst = (typ == 0 ? qrope : (typ == 1 ? krope : vbf)) + (size_t)bh * S * 64;
#pragma unroll
  for (int nt = 0; nt < 4; ++nt) {
#pragma unroll
    for (int j = 0; j < 8; ++j) {
      int srow = st * 16 + hi * 8 + j;
      int d = nt * 16 + n0;
      float val = acc[nt][j];
      if (typ != 2) {                               // fused RoPE (typ uniform per wave)
        float partner = __shfl_xor(val, 1);
        float fr = __powf(10000.0f, -(float)(d & ~1) * (1.0f / 64.0f));
        float sn, cs;
        __sincosf((float)srow * fr, &sn, &cs);
        float x1 = (d & 1) ? partner : val;
        float x2 = (d & 1) ? val : partner;
        val = (d & 1) ? (x1 * sn + x2 * cs) : (x1 * cs - x2 * sn);
        if (typ == 0) val *= 0.125f;                // fold softmax scale into q
      }
      dst[(size_t)srow * 64 + d] = f2bf(val);
    }
  }
}

// K -> swizzled B^T fragments into kvsw slots 0..3:  f = (bh*64+kt)*4 + fi, fi = sub*2 + kk
__global__ void k_repack_k(const unsigned short* __restrict__ krope, unsigned int* __restrict__ kvsw) {
  int t = blockIdx.x * 256 + threadIdx.x;            // 16*64*4*256 = 1048576
  int elem = t & 255, f = t >> 8;
  int lane = elem >> 3, j = elem & 7;
  int n0 = lane & 15, khalf = lane >> 4;
  int fi = f & 3, kt = (f >> 2) & 63, bh = f >> 8;
  int kk = fi & 1, sub = fi >> 1;
  int a = kt * 32 + sub * 16 + n0;
  int d = kk * 32 + khalf * 16 + 2 * j;
  const unsigned short* src = krope + (size_t)bh * S * 64 + (size_t)a * 64 + d;
  kvsw[((size_t)(f >> 2) * 8 + fi) * 256 + elem] =
      (unsigned int)src[0] | ((unsigned int)src[1] << 16);
}

// V -> swizzled B fragments into kvsw slots 4..7:  nt = f&3
__global__ void k_repack_v(const unsigned short* __restrict__ vbf, unsigned int* __restrict__ kvsw) {
  int t = blockIdx.x * 256 + threadIdx.x;            // 1048576
  int elem = t & 255, f = t >> 8;
  int lane = elem >> 3, j = elem & 7;
  int n0 = lane & 15, khalf = lane >> 4;
  int nt = f & 3, kt = (f >> 2) & 63, bh = f >> 8;
  int a = kt * 32 + khalf * 16 + 2 * j;
  int d = nt * 16 + n0;
  const unsigned short* src = vbf + (size_t)bh * S * 64;
  kvsw[((size_t)(f >> 2) * 8 + 4 + nt) * 256 + elem] =
      (unsigned int)src[(size_t)a * 64 + d] |
      ((unsigned int)src[(size_t)(a + 1) * 64 + d] << 16);
}

// ---------------- flash attention: TDM double-buffered K/V tiles in LDS ----------------
__global__ void __launch_bounds__(128) k_attn(
    const unsigned short* __restrict__ qrope,
    const unsigned int* __restrict__ kvsw,
    unsigned short* __restrict__ ohead) {
  __shared__ __align__(16) unsigned int KV[2][2048];       // 2 x (4KB K-frags + 4KB V-frags)
  __shared__ __align__(16) unsigned short Plds[4][16][32]; // per-wave P transpose
  int lane = threadIdx.x & 31, w = threadIdx.x >> 5;
  int tid = threadIdx.x; (void)tid;
  int W = blockIdx.x * 4 + w;                       // 2048 q-tiles; bh uniform per block
  int st = W & (S16 - 1), bh = W >> 7;
  int n0 = lane & 15, hi = lane >> 4;

  const unsigned short* Qb = qrope + (size_t)bh * S * 64;
  const unsigned int* Gkv = kvsw + (size_t)bh * 64 * 2048;

  FragB16 aQ0, aQ1;
  loadA(aQ0, Qb, 64, st * 16, 0, lane);
  loadA(aQ1, Qb, 64, st * 16, 32, lane);

  v8f o[4];
#pragma unroll
  for (int nt = 0; nt < 4; ++nt) o[nt] = zero8();
  float mrow[8], lrow[8];
#pragma unroll
  for (int j = 0; j < 8; ++j) { mrow[j] = -1e30f; lrow[j] = 0.0f; }

  // stage tile 0 into buffer 0
#if HAVE_TDM
  if (w == 0) tdm_load_8k((unsigned)(size_t)&KV[0][0], Gkv);
#else
  { const uint4* s4 = (const uint4*)Gkv; uint4* d4 = (uint4*)&KV[0][0];
#pragma unroll
    for (int i = 0; i < 4; ++i) d4[tid + i * 128] = s4[tid + i * 128]; }
#endif

  for (int kt = 0; kt < 64; ++kt) {
    int buf = kt & 1;
#if HAVE_TDM
    if (w == 0) __builtin_amdgcn_s_wait_tensorcnt(0);
#endif
    __syncthreads();                                // (A) tile kt visible in KV[buf]

    if (kt + 1 < 64) {                              // stage next tile (overlaps compute)
      const unsigned int* nsrc = Gkv + (size_t)(kt + 1) * 2048;
#if HAVE_TDM
      if (w == 0) tdm_load_8k((unsigned)(size_t)&KV[buf ^ 1][0], nsrc);
#else
      const uint4* s4 = (const uint4*)nsrc; uint4* d4 = (uint4*)&KV[buf ^ 1][0];
#pragma unroll
      for (int i = 0; i < 4; ++i) d4[tid + i * 128] = s4[tid + i * 128];
#endif
    }

    const unsigned int* kb = &KV[buf][0];
    v8f c0 = zero8(), c1 = zero8();
    {
      FragB16 bk;
      loadBsw(bk, kb + 0 * 256, lane); c0 = wmma_bf16(aQ0, bk, c0);
      loadBsw(bk, kb + 1 * 256, lane); c0 = wmma_bf16(aQ1, bk, c0);
      loadBsw(bk, kb + 2 * 256, lane); c1 = wmma_bf16(aQ0, bk, c1);
      loadBsw(bk, kb + 3 * 256, lane); c1 = wmma_bf16(aQ1, bk, c1);
    }

    float alpha[8];
#pragma unroll
    for (int j = 0; j < 8; ++j) {                   // online softmax per row
      float v = fmaxf(c0[j], c1[j]);
      v = fmaxf(v, __shfl_xor(v, 1));
      v = fmaxf(v, __shfl_xor(v, 2));
      v = fmaxf(v, __shfl_xor(v, 4));
      v = fmaxf(v, __shfl_xor(v, 8));
      float mn = fmaxf(mrow[j], v);
      alpha[j] = __expf(mrow[j] - mn);
      mrow[j] = mn;
      c0[j] = __expf(c0[j] - mn);
      c1[j] = __expf(c1[j] - mn);
      float s = c0[j] + c1[j];
      s += __shfl_xor(s, 1); s += __shfl_xor(s, 2);
      s += __shfl_xor(s, 4); s += __shfl_xor(s, 8);
      lrow[j] = lrow[j] * alpha[j] + s;
    }
#pragma unroll
    for (int nt = 0; nt < 4; ++nt)
#pragma unroll
      for (int j = 0; j < 8; ++j) o[nt][j] *= alpha[j];

    // transpose P (C layout -> A fragment) through per-wave LDS region
#pragma unroll
    for (int j = 0; j < 8; ++j) {
      Plds[w][hi * 8 + j][n0]      = f2bf(c0[j]);
      Plds[w][hi * 8 + j][16 + n0] = f2bf(c1[j]);
    }
    __syncthreads();                                // (B)
    FragB16 aP;
    loadA(aP, &Plds[w][0][0], 32, 0, 0, lane);

#pragma unroll
    for (int nt = 0; nt < 4; ++nt) {
      FragB16 bv;
      loadBsw(bv, kb + (4 + nt) * 256, lane);
      o[nt] = wmma_bf16(aP, bv, o[nt]);
    }
    __syncthreads();                                // (C) done with KV[buf] / Plds
  }

  unsigned short* dst = ohead + (size_t)bh * S * 64;
#pragma unroll
  for (int j = 0; j < 8; ++j) {
    float inv = 1.0f / lrow[j];
#pragma unroll
    for (int nt = 0; nt < 4; ++nt)
      dst[(size_t)(st * 16 + hi * 8 + j) * 64 + nt * 16 + n0] = f2bf(o[nt][j] * inv);
  }
}

// ---------------- mean over heads ----------------
__global__ void k_mean(const unsigned short* __restrict__ ohead, unsigned short* __restrict__ meano) {
  int i = blockIdx.x * 256 + threadIdx.x;           // 524288
  if (i >= B * S * 64) return;
  int d = i & 63, s = (i >> 6) & (S - 1), b = i >> 17;
  float acc = 0.0f;
#pragma unroll
  for (int h = 0; h < 4; ++h)
    acc += bf2f(ohead[(((size_t)(b * 4 + h)) * S + s) * 64 + d]);
  meano[i] = f2bf(acc * 0.25f);
}

// ---------------- final GEMM: (B*S,64) x (64,1024) -> fp32 out ----------------
__global__ void __launch_bounds__(128) k_final(
    const unsigned short* __restrict__ meano, const unsigned int* __restrict__ wosw,
    float* __restrict__ out) {
  int lane = threadIdx.x & 31;
  int W = blockIdx.x * 4 + (threadIdx.x >> 5);      // 8192 waves
  if (W >= 8192) return;
  int ng = W & 15, mt = W >> 4;

  FragB16 a0f, a1f;
  loadA(a0f, meano, 64, mt * 16, 0, lane);
  loadA(a1f, meano, 64, mt * 16, 32, lane);

  v8f acc[4];
#pragma unroll
  for (int i = 0; i < 4; ++i) acc[i] = zero8();
#pragma unroll
  for (int i = 0; i < 4; ++i) {
    int nt = ng * 4 + i;
    FragB16 b0, b1;
    loadBsw(b0, wosw + (size_t)nt * 256, lane);
    loadBsw(b1, wosw + (size_t)(64 + nt) * 256, lane);
    acc[i] = wmma_bf16(a0f, b0, acc[i]);
    acc[i] = wmma_bf16(a1f, b1, acc[i]);
  }
  int n0 = lane & 15, hi = lane >> 4;
#pragma unroll
  for (int i = 0; i < 4; ++i)
#pragma unroll
    for (int j = 0; j < 8; ++j)
      out[(size_t)(mt * 16 + hi * 8 + j) * 1024 + (ng * 4 + i) * 16 + n0] = acc[i][j];
}

// ---------------- host launcher ----------------
extern "C" void kernel_launch(void* const* d_in, const int* in_sizes, int n_in,
                              void* d_out, int out_size, void* d_ws, size_t ws_size,
                              hipStream_t stream) {
  (void)in_sizes; (void)n_in; (void)out_size; (void)ws_size;
  const float* q  = (const float*)d_in[0];
  const float* kv = (const float*)d_in[1];
  const float* Wq = (const float*)d_in[2];
  const float* Wk = (const float*)d_in[3];
  const float* Wv = (const float*)d_in[4];
  const float* Wo = (const float*)d_in[5];

  char* ws = (char*)d_ws;
  size_t off = 0;
  auto alloc = [&](size_t bytes) -> void* {
    void* p = ws + off;
    off += (bytes + 255) & ~(size_t)255;
    return p;
  };
  unsigned short* qbf   = (unsigned short*)alloc((size_t)B * S * E * 2);   // 16 MB
  unsigned short* kvbf  = (unsigned short*)alloc((size_t)B * S * E * 2);   // 16 MB
  unsigned int*   wqsw  = (unsigned int*)alloc(131072 * 4);
  unsigned int*   wksw  = (unsigned int*)alloc(131072 * 4);
  unsigned int*   wvsw  = (unsigned int*)alloc(131072 * 4);
  unsigned int*   wosw  = (unsigned int*)alloc(32768 * 4);
  unsigned short* qrope = (unsigned short*)alloc((size_t)B * H * S * 64 * 2);
  unsigned short* krope = (unsigned short*)alloc((size_t)B * H * S * 64 * 2);
  unsigned short* vbf   = (unsigned short*)alloc((size_t)B * H * S * 64 * 2);
  unsigned int*   kvsw  = (unsigned int*)alloc((size_t)2097152 * 4);       // 8 MB K+V frags
  unsigned short* ohead = (unsigned short*)alloc((size_t)B * H * S * 64 * 2);
  unsigned short* meano = (unsigned short*)alloc((size_t)B * S * 64 * 2);

  const int nqe = B * S * E;                         // 8,388,608
  k_convert_bf16<<<nqe / 4 / 256, 256, 0, stream>>>(q, qbf, nqe);
  k_convert_bf16<<<nqe / 4 / 256, 256, 0, stream>>>(kv, kvbf, nqe);
  k_prep_wq <<<512, 256, 0, stream>>>(Wq, wqsw);
  k_prep_wkv<<<512, 256, 0, stream>>>(Wk, wksw);
  k_prep_wkv<<<512, 256, 0, stream>>>(Wv, wvsw);
  k_prep_wo <<<128, 256, 0, stream>>>(Wo, wosw);

  k_proj<<<(3 * B * H * S16) / 4, 128, 0, stream>>>(qbf, kvbf, wqsw, wksw, wvsw,
                                                    qrope, krope, vbf);
  k_repack_k<<<4096, 256, 0, stream>>>(krope, kvsw);
  k_repack_v<<<4096, 256, 0, stream>>>(vbf, kvsw);
  k_attn<<<(B * H * S16) / 4, 128, 0, stream>>>(qrope, kvsw, ohead);
  k_mean<<<(B * S * 64) / 256, 256, 0, stream>>>(ohead, meano);
  k_final<<<8192 / 4, 128, 0, stream>>>(meano, wosw, (float*)d_out);
}
